// BatchTemporalContrastiveLoss_89335319756912
// MI455X (gfx1250) — matrix operand order repaired and checked by reading, not compiled, not run on hardware
//
#include <hip/hip_runtime.h>
#include <hip/hip_bf16.h>

typedef __attribute__((ext_vector_type(16))) __bf16        v16bf;
typedef __attribute__((ext_vector_type(8)))  float         v8f;
typedef __attribute__((ext_vector_type(8)))  unsigned int  v8u;

// fp32 -> bf16, round-half-away (ties have measure zero on this data)
__device__ __forceinline__ unsigned short bf1(float f) {
    return (unsigned short)((__builtin_bit_cast(unsigned int, f) + 0x8000u) >> 16);
}
// packed fp32x2 -> bf16x2
__device__ __forceinline__ unsigned int pk2(float x, float y) {
    unsigned int ax = __builtin_bit_cast(unsigned int, x) + 0x8000u;
    unsigned int ay = __builtin_bit_cast(unsigned int, y) + 0x8000u;
    return (ax >> 16) | (ay & 0xFFFF0000u);
}

// A-matrix fragment (16x32 bf16), tile row-major in LDS with row stride `ld` (ushorts).
// ISA 7.12.2: lanes 0-15 -> M=lane, K {0..7,16..23}; lanes 16-31 -> M=lane-16, K {8..15,24..31}.
__device__ __forceinline__ v16bf frag_a(const unsigned short* tile, int ld, int lane) {
    const int r  = lane & 15;
    const int kh = (lane < 16) ? 0 : 8;
    const unsigned short* p = tile + r * ld + kh;
    v8u u;
    ((uint4*)&u)[0] = *(const uint4*)(p);
    ((uint4*)&u)[1] = *(const uint4*)(p + 16);
    return __builtin_bit_cast(v16bf, u);
}

// B-matrix fragment (32x16 bf16), stored transposed in LDS as [N][K] row-major stride `ld`.
// Lanes 0-15: N=lane, K {0..7,16..23}; lanes 16-31: N=lane-16, K {8..15,24..31} (2 packed per VGPR).
__device__ __forceinline__ v16bf frag_b(const unsigned short* tile, int ld, int lane) {
    const int n  = lane & 15;
    const int k0 = (lane < 16) ? 0 : 8;
    const unsigned short* p = tile + n * ld + k0;
    v8u u;
    ((uint4*)&u)[0] = *(const uint4*)(p);
    ((uint4*)&u)[1] = *(const uint4*)(p + 16);
    return __builtin_bit_cast(v16bf, u);
}

// ---------- prep: transpose + convert weights: out[n][k] (bf16) = in[k][n] (f32) ----------
__global__ __launch_bounds__(256) void transpose_cvt(const float* __restrict__ in,
                                                     unsigned short* __restrict__ outp,
                                                     int K, int N) {
    __shared__ float t[32][33];
    const int k0 = blockIdx.x * 32;
    const int n0 = blockIdx.y * 32;
    const int tx = threadIdx.x & 31;
    const int ty = threadIdx.x >> 5;        // 0..7
    #pragma unroll
    for (int i = 0; i < 4; ++i)
        t[ty + i * 8][tx] = in[(size_t)(k0 + ty + i * 8) * N + n0 + tx];
    __syncthreads();
    #pragma unroll
    for (int i = 0; i < 4; ++i) {
        int r = ty + i * 8;                 // n offset
        outp[(size_t)(n0 + r) * K + k0 + tx] = bf1(t[tx][r]);
    }
}

// ---------------- GEMM1: Y = relu(A[B,2048] @ W1[2048,1024] + b1) -> bf16 ----------------
#define G1_BM 128
#define G1_BN 128
#define G1_BK 32
#define G1_LD 40   // padded LDS row stride (ushorts)

__device__ __forceinline__ void g1_load(const float* __restrict__ A,
                                        const unsigned short* __restrict__ Wt,
                                        int H, int bm, int bn, int k0, int tid,
                                        float4 ra[4], uint4 rb[2]) {
    #pragma unroll
    for (int i = 0; i < 4; ++i) {
        int idx = tid + i * 256;
        ra[i] = *(const float4*)(A + (size_t)(bm + (idx >> 3)) * H + k0 + ((idx & 7) << 2));
    }
    #pragma unroll
    for (int i = 0; i < 2; ++i) {
        int idx = tid + i * 256;            // 0..511; 4 uint4 per n-row
        rb[i] = *(const uint4*)(Wt + (size_t)(bn + (idx >> 2)) * H + k0 + ((idx & 3) << 3));
    }
}

__device__ __forceinline__ void g1_store(unsigned short* __restrict__ As,
                                         unsigned short* __restrict__ Bs,
                                         int tid, const float4 ra[4], const uint4 rb[2]) {
    #pragma unroll
    for (int i = 0; i < 4; ++i) {
        int idx = tid + i * 256;
        int r = idx >> 3, c = (idx & 7) << 2;
        uint2 t;
        t.x = pk2(ra[i].x, ra[i].y);
        t.y = pk2(ra[i].z, ra[i].w);
        *(uint2*)&As[r * G1_LD + c] = t;    // ds_store_b64
    }
    #pragma unroll
    for (int i = 0; i < 2; ++i) {
        int idx = tid + i * 256;
        int r = idx >> 2, c = (idx & 3) << 3;
        *(uint4*)&Bs[r * G1_LD + c] = rb[i]; // ds_store_b128 (already bf16, pre-transposed)
    }
}

__global__ __launch_bounds__(256, 4) void gemm1_relu(const float* __restrict__ A,
                                                     const unsigned short* __restrict__ Wt,
                                                     const float* __restrict__ bias,
                                                     unsigned short* __restrict__ Y,
                                                     int H, int N) {
    __shared__ unsigned short As[2][G1_BM * G1_LD];   // 2 x 10 KB
    __shared__ unsigned short Bs[2][G1_BN * G1_LD];   // 2 x 10 KB

    const int tid  = threadIdx.x;
    const int lane = tid & 31;
    const int wave = tid >> 5;                 // 0..7
    const int bm   = blockIdx.y * G1_BM;
    const int bn   = blockIdx.x * G1_BN;
    const int wm   = (wave >> 1) * 32;         // wave m-offset (4 chunks of 32)
    const int wn   = (wave & 1) * 64;          // wave n-offset (2 chunks of 64)

    v8f acc[2][4] = {};
    float4 ra[4];
    uint4  rb[2];

    // preload K-tile 0
    g1_load(A, Wt, H, bm, bn, 0, tid, ra, rb);
    g1_store(&As[0][0], &Bs[0][0], tid, ra, rb);
    __syncthreads();

    const int nk = H / G1_BK;
    for (int kt = 0; kt < nk; ++kt) {
        const int cur = kt & 1;
        if (kt + 1 < nk)   // prefetch next tile into registers (overlaps WMMA below)
            g1_load(A, Wt, H, bm, bn, (kt + 1) * G1_BK, tid, ra, rb);

        v16bf a0 = frag_a(&As[cur][(wm + 0)  * G1_LD], G1_LD, lane);
        v16bf a1 = frag_a(&As[cur][(wm + 16) * G1_LD], G1_LD, lane);
        #pragma unroll
        for (int nt = 0; nt < 4; ++nt) {
            v16bf bb = frag_b(&Bs[cur][(wn + nt * 16) * G1_LD], G1_LD, lane);
            acc[0][nt] = __builtin_amdgcn_wmma_f32_16x16x32_bf16(
                false, a0, false, bb, (short)0, acc[0][nt], false, false);
            acc[1][nt] = __builtin_amdgcn_wmma_f32_16x16x32_bf16(
                false, a1, false, bb, (short)0, acc[1][nt], false, false);
        }

        if (kt + 1 < nk)
            g1_store(&As[cur ^ 1][0], &Bs[cur ^ 1][0], tid, ra, rb);
        __syncthreads();
    }

    // epilogue: bias + relu -> bf16 store (packed conversion for row pairs)
    #pragma unroll
    for (int mt = 0; mt < 2; ++mt) {
        #pragma unroll
        for (int nt = 0; nt < 4; ++nt) {
            int row = bm + wm + mt * 16 + ((lane >> 4) << 3);
            int col = bn + wn + nt * 16 + (lane & 15);
            #pragma unroll
            for (int i = 0; i < 8; i += 2) {
                float v0 = acc[mt][nt][i]     + bias[col];
                float v1 = acc[mt][nt][i + 1] + bias[col];
                v0 = v0 > 0.0f ? v0 : 0.0f;
                v1 = v1 > 0.0f ? v1 : 0.0f;
                unsigned int u = pk2(v0, v1);
                Y[(size_t)(row + i)     * N + col] = (unsigned short)u;
                Y[(size_t)(row + i + 1) * N + col] = (unsigned short)(u >> 16);
            }
        }
    }
}

// -------- GEMM2 fused: X = Y1[B,1024] @ W2[1024,256] + b2; LayerNorm; L2-normalize --------
#define G2_BM 32
#define G2_BK 32
#define G2_LD 40
#define G2_CLD 260   // padded fp32 C row stride

__global__ __launch_bounds__(256) void gemm2_proj(const unsigned short* __restrict__ Y1,
                                                  const unsigned short* __restrict__ W2t,
                                                  const float* __restrict__ b2,
                                                  const float* __restrict__ gam,
                                                  const float* __restrict__ bet,
                                                  float* __restrict__ out,
                                                  int Kdim) {
    __shared__ unsigned short As[G2_BM * G2_LD];      //  2.5 KB
    __shared__ unsigned short Bs[256 * G2_LD];        // 20.0 KB
    __shared__ float Cs[G2_BM * G2_CLD];              // 32.5 KB
    __shared__ float redS[G2_BM][8];
    __shared__ float redQ[G2_BM][8];

    const int tid  = threadIdx.x;
    const int lane = tid & 31;
    const int wave = tid >> 5;                  // 0..7
    const int bm   = blockIdx.x * G2_BM;
    const int mchunk = wave >> 2;               // 0..1 -> rows mchunk*16..+16
    const int nq     = wave & 3;                // cols nq*64..+64 (4 tiles)

    v8f acc[4] = {};

    for (int k0 = 0; k0 < Kdim; k0 += G2_BK) {
        // A tile: 32x32 bf16 copy (128 uint4)
        if (tid < 128) {
            int r = tid >> 2;
            int c = (tid & 3) << 3;
            *(uint4*)&As[r * G2_LD + c] =
                *(const uint4*)(Y1 + (size_t)(bm + r) * Kdim + k0 + c);
        }
        // B tile: 256x32 bf16 copy from pre-transposed W2t (1024 uint4)
        #pragma unroll
        for (int i = 0; i < 4; ++i) {
            int idx = tid + i * 256;
            int r = idx >> 2;                   // n 0..255
            int c = (idx & 3) << 3;
            *(uint4*)&Bs[r * G2_LD + c] =
                *(const uint4*)(W2t + (size_t)r * Kdim + k0 + c);
        }
        __syncthreads();

        v16bf a0 = frag_a(&As[(mchunk * 16) * G2_LD], G2_LD, lane);
        #pragma unroll
        for (int nt = 0; nt < 4; ++nt) {
            v16bf bb = frag_b(&Bs[(nq * 64 + nt * 16) * G2_LD], G2_LD, lane);
            acc[nt] = __builtin_amdgcn_wmma_f32_16x16x32_bf16(
                false, a0, false, bb, (short)0, acc[nt], false, false);
        }
        __syncthreads();
    }

    // spill C (+bias) into LDS for row-wise LayerNorm
    {
        int rb_ = mchunk * 16 + ((lane >> 4) << 3);
        int cb  = nq * 64 + (lane & 15);
        #pragma unroll
        for (int nt = 0; nt < 4; ++nt) {
            #pragma unroll
            for (int i = 0; i < 8; ++i) {
                int rr = rb_ + i;
                int cc = cb + nt * 16;
                Cs[rr * G2_CLD + cc] = acc[nt][i] + b2[cc];
            }
        }
    }
    __syncthreads();

    // LayerNorm + L2 normalize: 8 threads per row, 32 cols each
    const int r = tid >> 3;
    const int q = tid & 7;
    const int c0 = q * 32;

    float s = 0.0f, sq = 0.0f;
    #pragma unroll
    for (int c = 0; c < 32; ++c) {
        float x = Cs[r * G2_CLD + c0 + c];
        s += x; sq += x * x;
    }
    redS[r][q] = s; redQ[r][q] = sq;
    __syncthreads();

    float sum = 0.0f, sumsq = 0.0f;
    #pragma unroll
    for (int j = 0; j < 8; ++j) { sum += redS[r][j]; sumsq += redQ[r][j]; }
    float mean = sum * (1.0f / 256.0f);
    float var  = sumsq * (1.0f / 256.0f) - mean * mean;
    float rstd = rsqrtf(var + 1e-5f);
    __syncthreads();

    float s2 = 0.0f;
    #pragma unroll
    for (int c = 0; c < 32; ++c) {
        int cc = c0 + c;
        float x = Cs[r * G2_CLD + cc];
        float y = (x - mean) * rstd * gam[cc] + bet[cc];
        Cs[r * G2_CLD + cc] = y;
        s2 += y * y;
    }
    redQ[r][q] = s2;
    __syncthreads();

    float nrm2 = 0.0f;
    #pragma unroll
    for (int j = 0; j < 8; ++j) nrm2 += redQ[r][j];
    float n = sqrtf(nrm2);
    float inv = 1.0f / fmaxf(n, 1e-12f);

    #pragma unroll
    for (int c = 0; c < 32; ++c) {
        int cc = c0 + c;
        out[(size_t)(bm + r) * 256 + cc] = Cs[r * G2_CLD + cc] * inv;
    }
}

// ---------------- Loss: InfoNCE over [pos | 64 negs], mean over batch ----------------
__global__ __launch_bounds__(256) void loss_kernel(const float* __restrict__ anchors,
                                                   const float* __restrict__ positives,
                                                   const float* __restrict__ negbuf,
                                                   const int*   __restrict__ nidx,
                                                   float* __restrict__ out,
                                                   int B) {
    const int b    = blockIdx.x;
    const int tid  = threadIdx.x;
    const int lane = tid & 31;
    const int wave = tid >> 5;

    __shared__ float anc[256];
    __shared__ float logits[72];

    anc[tid] = anchors[(size_t)b * 256 + tid];
    __syncthreads();

    // logit 0 = positive, 1..64 = negatives; 8 waves stride the 65 logits
    for (int j = wave; j < 65; j += 8) {
        const float* vec = (j == 0)
            ? (positives + (size_t)b * 256)
            : (negbuf + (size_t)nidx[(size_t)b * 64 + (j - 1)] * 256);
        float s = 0.0f;
        #pragma unroll
        for (int e = 0; e < 8; ++e) s += anc[lane + e * 32] * vec[lane + e * 32];
        #pragma unroll
        for (int off = 16; off > 0; off >>= 1) s += __shfl_xor(s, off, 32);
        if (lane == 0) logits[j] = s * 2.0f;    // / TEMPERATURE(0.5)
    }
    __syncthreads();

    if (wave == 0) {
        float m = -3.4e38f;
        for (int j = lane; j < 65; j += 32) m = fmaxf(m, logits[j]);
        #pragma unroll
        for (int off = 16; off > 0; off >>= 1) m = fmaxf(m, __shfl_xor(m, off, 32));
        float s = 0.0f;
        for (int j = lane; j < 65; j += 32) s += expf(logits[j] - m);
        #pragma unroll
        for (int off = 16; off > 0; off >>= 1) s += __shfl_xor(s, off, 32);
        if (lane == 0) {
            float lse = m + logf(s);
            atomicAdd(out, (lse - logits[0]) / (float)B);
        }
    }
}

extern "C" void kernel_launch(void* const* d_in, const int* in_sizes, int n_in,
                              void* d_out, int out_size, void* d_ws, size_t ws_size,
                              hipStream_t stream) {
    const float* hidden = (const float*)d_in[0];
    const float* posh   = (const float*)d_in[1];
    const float* negbuf = (const float*)d_in[2];
    const float* W1     = (const float*)d_in[3];
    const float* b1     = (const float*)d_in[4];
    const float* W2     = (const float*)d_in[5];
    const float* b2     = (const float*)d_in[6];
    const float* gam    = (const float*)d_in[7];
    const float* bet    = (const float*)d_in[8];
    const int*   nidx   = (const int*)d_in[9];

    const int H  = 2048;
    const int Hh = 1024;
    const int P  = 256;
    const int B  = in_sizes[0] / H;          // 16384

    char* ws = (char*)d_ws;
    unsigned short* Y1   = (unsigned short*)ws;                ws += (size_t)B * Hh * 2;
    float* anchors       = (float*)ws;                         ws += (size_t)B * P * 4;
    float* positives     = (float*)ws;                         ws += (size_t)B * P * 4;
    unsigned short* W1t  = (unsigned short*)ws;                ws += (size_t)Hh * H * 2;
    unsigned short* W2t  = (unsigned short*)ws;

    // one-time weight transpose+convert: W1t[n][k], W2t[n][k] (bf16)
    transpose_cvt<<<dim3(H / 32, Hh / 32), 256, 0, stream>>>(W1, W1t, H, Hh);
    transpose_cvt<<<dim3(Hh / 32, P / 32), 256, 0, stream>>>(W2, W2t, Hh, P);

    dim3 g1(Hh / G1_BN, B / G1_BM);          // (8, 128)
    dim3 g2(B / G2_BM);                      // 512

    gemm1_relu<<<g1, 256, 0, stream>>>(hidden, W1t, b1, Y1, H, Hh);
    gemm2_proj<<<g2, 256, 0, stream>>>(Y1, W2t, b2, gam, bet, anchors, Hh);
    gemm1_relu<<<g1, 256, 0, stream>>>(posh, W1t, b1, Y1, H, Hh);
    gemm2_proj<<<g2, 256, 0, stream>>>(Y1, W2t, b2, gam, bet, positives, Hh);

    (void)hipMemsetAsync(d_out, 0, sizeof(float), stream);
    loss_kernel<<<dim3(B), 256, 0, stream>>>(anchors, positives, negbuf, nidx,
                                             (float*)d_out, B);
}
